// linear_attention_2147483648258
// MI455X (gfx1250) — matrix operand order repaired and checked by reading, not compiled, not run on hardware
//
#include <hip/hip_runtime.h>
#include <hip/hip_bf16.h>

// ---------------------------------------------------------------------------
// Linear attention (elu+1 feature map) for MI455X / gfx1250.
// All GEMMs use v_wmma_f32_16x16x32_bf16 (f32 accumulate). Big GEMMs stage
// the shared A-band through LDS with GLOBAL_LOAD_ASYNC_TO_LDS_B128
// (double-buffered, s_wait_asynccnt pipelined) when the builtin exists.
// ---------------------------------------------------------------------------

typedef __attribute__((ext_vector_type(16))) __bf16 v16bf;
typedef __attribute__((ext_vector_type(8)))  __bf16 v8bf;
typedef __attribute__((ext_vector_type(8)))  float  v8f;

// Async-copy operand types: the gfx1250 builtin takes pointers to 16B int
// vectors in AS1 (global) / AS3 (LDS).
typedef int v4i_vs __attribute__((vector_size(16)));
typedef __attribute__((address_space(1))) v4i_vs as1_v4i;
typedef __attribute__((address_space(3))) v4i_vs as3_v4i;

#define NB      4
#define SEQ     4096
#define DMODEL  1024
#define NHEADS  16
#define DHEAD   64
#define EPSF    1e-6f

#if __has_builtin(__builtin_amdgcn_global_load_async_to_lds_b128) && \
    __has_builtin(__builtin_amdgcn_s_wait_asynccnt)
#define HAVE_ASYNC_LDS 1
#else
#define HAVE_ASYNC_LDS 0
#endif

// ---- WMMA wrapper ---------------------------------------------------------
__device__ inline v8f wmma_bf16(v16bf a, v16bf b, v8f c) {
  // (neg_a, A, neg_b, B, c_mod, C, reuse_a, reuse_b)
  return __builtin_amdgcn_wmma_f32_16x16x32_bf16(false, a, false, b,
                                                 (short)0, c, false, false);
}

// A fragment (16x32): lane holds 8 contiguous K at +0 and 8 at +16 (elements).
// p is 16B-aligned.  bf16 source, vectorized.
__device__ inline v16bf load_a16(const __bf16* p) {
  v8bf lo = *(const v8bf*)p;
  v8bf hi = *(const v8bf*)(p + 16);
  v16bf r;
#pragma unroll
  for (int i = 0; i < 8; i++) { r[i] = lo[i]; r[8 + i] = hi[i]; }
  return r;
}
// B fragment (32x16): lane holds 8 contiguous K at +0 and 8 at +16 (16B-aligned).
__device__ inline v16bf load_b16(const __bf16* p) {
  v8bf lo = *(const v8bf*)p;
  v8bf hi = *(const v8bf*)(p + 16);
  v16bf r;
#pragma unroll
  for (int i = 0; i < 8; i++) { r[i] = lo[i]; r[8 + i] = hi[i]; }
  return r;
}

// ---------------------------------------------------------------------------
// Weight conversion: f32 -> bf16 (one pass, 1M elements per matrix)
// ---------------------------------------------------------------------------
__global__ void cvt_f32_bf16(const float* __restrict__ src,
                             __bf16* __restrict__ dst, int n) {
  int i = (blockIdx.x * blockDim.x + threadIdx.x) * 4;
  if (i < n) {
#pragma unroll
    for (int j = 0; j < 4; j++) dst[i + j] = (__bf16)src[i + j];
  }
}

// ---------------------------------------------------------------------------
// GEMM: out[r,c] = act( sum_k X[r,k] * W[c,k] + bias[c] ), W pre-converted bf16.
// Block = 8 waves, block tile 64(M, shared) x 512(N); wave tile 64x64.
// A-band (64 x 32 chunk of X) staged into LDS (async copy when available),
// double-buffered.  TX in {float,__bf16}; TOUT in {float,__bf16}.
// ---------------------------------------------------------------------------
template <typename TX, bool ELU1, typename TOUT>
__global__ void __launch_bounds__(256)
gemm_xwT(const TX* __restrict__ X, const __bf16* __restrict__ Wb,
         const float* __restrict__ bias, TOUT* __restrict__ out,
         int M, int NC, int K) {
  constexpr int APAD = (sizeof(TX) == 4) ? 36 : 40;  // padded row (elements)
  __shared__ __align__(16) TX As[2][64 * APAD];

  const int tid  = threadIdx.x;
  const int lane = tid & 31, l15 = lane & 15, hf = lane >> 4;
  const int wv   = tid >> 5;

  const int nbBlocks = NC >> 9;                                  // NC/512
  const int mb = (blockIdx.x / nbBlocks) << 6;
  const int nb = ((blockIdx.x % nbBlocks) << 9) + (wv << 6);

  constexpr int ELEM_PER_SEG = 16 / sizeof(TX);      // elements per 16B
  constexpr int SEG_PER_ROW  = 32 / ELEM_PER_SEG;    // segs per 32-elem row
  constexpr int TOTAL_SEG    = 64 * SEG_PER_ROW;
  constexpr int SEG_PER_THR  = TOTAL_SEG / 256;      // 2 (f32) or 1 (bf16)

  auto stage = [&](int buf, int kb) {
#pragma unroll
    for (int s0 = 0; s0 < SEG_PER_THR; s0++) {
      const int s   = tid + s0 * 256;
      const int row = s / SEG_PER_ROW;
      const int c   = (s % SEG_PER_ROW) * ELEM_PER_SEG;
      const TX* gp  = X + (size_t)(mb + row) * K + kb + c;
      TX*       lp  = &As[buf][row * APAD + c];
#if HAVE_ASYNC_LDS
      __builtin_amdgcn_global_load_async_to_lds_b128(
          (as1_v4i*)(uintptr_t)gp, (as3_v4i*)(uint32_t)(uintptr_t)lp, 0, 0);
#else
      *(uint4*)lp = *(const uint4*)gp;   // 16B sync copy fallback
#endif
    }
  };

  const v8f vz = {0.f, 0.f, 0.f, 0.f, 0.f, 0.f, 0.f, 0.f};
  v8f acc[4][4];
#pragma unroll
  for (int i = 0; i < 4; i++)
#pragma unroll
    for (int j = 0; j < 4; j++) acc[i][j] = vz;

  const int nIter = K >> 5;
  stage(0, 0);

  for (int it = 0; it < nIter; ++it) {
    const int buf = it & 1;
    const bool pre = (it + 1) < nIter;
    if (pre) stage(buf ^ 1, (it + 1) << 5);
#if HAVE_ASYNC_LDS
    if (pre) __builtin_amdgcn_s_wait_asynccnt(SEG_PER_THR);  // current buf done
    else     __builtin_amdgcn_s_wait_asynccnt(0);
#endif
    __syncthreads();

    const int kc = (it << 5) + hf * 8;  // global k for B fragments
    v16bf a[4], b[4];
    const TX* Ab = &As[buf][0];
#pragma unroll
    for (int mi = 0; mi < 4; mi++) {
      const TX* p = Ab + (mi * 16 + l15) * APAD + hf * 8;
      if constexpr (sizeof(TX) == 2) {
        a[mi] = load_a16((const __bf16*)p);
      } else {
        v16bf r;
#pragma unroll
        for (int i = 0; i < 8; i++) {
          r[i]     = (__bf16)(float)p[i];
          r[8 + i] = (__bf16)(float)p[16 + i];
        }
        a[mi] = r;
      }
    }
#pragma unroll
    for (int ni = 0; ni < 4; ni++) {
      const __bf16* p = Wb + (size_t)(nb + ni * 16 + l15) * K + kc;
      b[ni] = load_b16(p);
      if (pre) __builtin_prefetch(p + 32, 0, 0);   // next k-chunk of weights
    }
#pragma unroll
    for (int mi = 0; mi < 4; mi++)
#pragma unroll
      for (int ni = 0; ni < 4; ni++)
        acc[mi][ni] = wmma_bf16(a[mi], b[ni], acc[mi][ni]);

    __syncthreads();
  }

  // Epilogue. C layout: VGPR r -> row = r + half*8, col = lane&15.
#pragma unroll
  for (int ni = 0; ni < 4; ni++) {
    const int col = nb + ni * 16 + l15;
    const float bc = bias[col];
#pragma unroll
    for (int mi = 0; mi < 4; mi++) {
#pragma unroll
      for (int r = 0; r < 8; r++) {
        const int row = mb + mi * 16 + hf * 8 + r;
        float v = acc[mi][ni][r] + bc;
        if constexpr (ELU1) v = (v > 0.f) ? (v + 1.f) : __expf(v);
        out[(size_t)row * NC + col] = (TOUT)v;
      }
    }
  }
}

// ---------------------------------------------------------------------------
// ksum[n, c] = sum_s kf[n, s, c]   (c = h*64 + d), chunked with f32 atomics
// ---------------------------------------------------------------------------
__global__ void zero_f32(float* __restrict__ p, int n) {
  int i = blockIdx.x * blockDim.x + threadIdx.x;
  if (i < n) p[i] = 0.f;
}

__global__ void ksum_kernel(const __bf16* __restrict__ kf,
                            float* __restrict__ ksum) {
  const int CH = 16, ROWS = SEQ / 16;
  const int t  = blockIdx.x * blockDim.x + threadIdx.x;
  const int c  = t % DMODEL;
  const int r  = t / DMODEL;
  const int ch = r % CH;
  const int n  = r / CH;
  const __bf16* p = kf + ((size_t)n * SEQ + (size_t)ch * ROWS) * DMODEL + c;
  float s = 0.f;
  for (int i = 0; i < ROWS; i++) s += (float)p[(size_t)i * DMODEL];
  atomicAdd(&ksum[n * DMODEL + c], s);
}

// ---------------------------------------------------------------------------
// KV[n,h,m,d] = sum_s vb[n,s,h,m] * kf[n,s,h,d]
// One block per (n,h); 32-wide s-chunks staged through LDS with transpose so
// both WMMA operands become s-contiguous. Row pad 40 -> 16B-aligned fragment
// reads, bank-stride 20 (conflict-free over 16 lanes).
// ---------------------------------------------------------------------------
__global__ void __launch_bounds__(128)
kv_kernel(const __bf16* __restrict__ kf, const __bf16* __restrict__ vb,
          __bf16* __restrict__ KVb) {
  __shared__ __align__(16) __bf16 Kt[64][40];  // [d][s]
  __shared__ __align__(16) __bf16 Vt[64][40];  // [m][s]
  const int n = blockIdx.x >> 4, h = blockIdx.x & 15;
  const int tid = threadIdx.x;
  const int wv = tid >> 5, lane = tid & 31, l15 = lane & 15, hf = lane >> 4;
  const size_t base = (size_t)n * SEQ * DMODEL + (size_t)h * DHEAD;

  const v8f vz = {0.f, 0.f, 0.f, 0.f, 0.f, 0.f, 0.f, 0.f};
  v8f acc[4];
#pragma unroll
  for (int i = 0; i < 4; i++) acc[i] = vz;

  const int sl = tid >> 2;         // staged s row 0..31
  const int dg = (tid & 3) << 4;   // staged channel group 0,16,32,48

  for (int sb = 0; sb < SEQ; sb += 32) {
    __bf16 kr[16], vr[16];
    const __bf16* kp = kf + base + (size_t)(sb + sl) * DMODEL + dg;
    const __bf16* vp = vb + base + (size_t)(sb + sl) * DMODEL + dg;
#pragma unroll
    for (int i = 0; i < 16; i++) { kr[i] = kp[i]; vr[i] = vp[i]; }
    __syncthreads();               // previous iteration's LDS reads done
#pragma unroll
    for (int i = 0; i < 16; i++) { Kt[dg + i][sl] = kr[i]; Vt[dg + i][sl] = vr[i]; }
    __syncthreads();

    // A = v^T (m x s), B = K^T rows: lane j reads Kt[j][s..] contiguously
    const int m = (wv << 4) + l15;
    v16bf a = load_a16(&Vt[m][hf * 8]);
#pragma unroll
    for (int dt = 0; dt < 4; dt++) {
      v16bf b = load_b16(&Kt[(dt << 4) + l15][hf * 8]);
      acc[dt] = wmma_bf16(a, b, acc[dt]);
    }
  }

  __bf16* outp = KVb + (size_t)blockIdx.x * DHEAD * DHEAD;  // [m][d] row-major
#pragma unroll
  for (int dt = 0; dt < 4; dt++)
#pragma unroll
    for (int r = 0; r < 8; r++) {
      const int m = (wv << 4) + hf * 8 + r;
      const int d = (dt << 4) + l15;
      outp[m * DHEAD + d] = (__bf16)acc[dt][r];
    }
}

// ---------------------------------------------------------------------------
// Z[n,l,h] = 1 / (sum_d qf[n,l,h*64+d] * ksum[n,h*64+d] + eps)
// ---------------------------------------------------------------------------
__global__ void z_kernel(const __bf16* __restrict__ qf,
                         const float* __restrict__ ksum,
                         float* __restrict__ Z) {
  const int t  = blockIdx.x * blockDim.x + threadIdx.x;  // n*L*H + l*H + h
  const int h  = t % NHEADS;
  const int nl = t / NHEADS;
  const int l  = nl % SEQ;
  const int n  = nl / SEQ;
  const __bf16* q  = qf + ((size_t)n * SEQ + l) * DMODEL + h * DHEAD;
  const float*  ks = ksum + n * DMODEL + h * DHEAD;
  float s = 0.f;
#pragma unroll 8
  for (int d = 0; d < DHEAD; d++) s += (float)q[d] * ks[d];
  Z[t] = 1.f / (s + EPSF);
}

// ---------------------------------------------------------------------------
// Vh[n,l,h*64+m] = Z[n,l,h] * sum_d qf[n,l,h*64+d] * KV[n,h,m,d]
// One wave per (n,h, 64-row l-tile); KV is tiny and L2-resident.
// ---------------------------------------------------------------------------
__global__ void __launch_bounds__(256)
attn_kernel(const __bf16* __restrict__ qf, const __bf16* __restrict__ KVb,
            const float* __restrict__ Z, __bf16* __restrict__ Vh) {
  const int lane = threadIdx.x & 31, l15 = lane & 15, hf = lane >> 4;
  const int wid  = blockIdx.x * (blockDim.x >> 5) + (threadIdx.x >> 5);
  const int lt   = wid & 63;       // l tile (SEQ/64 = 64)
  const int nh   = wid >> 6;
  const int n    = nh >> 4, h = nh & 15;

  const __bf16* Q = qf + (size_t)n * SEQ * DMODEL + (size_t)h * DHEAD;
  const __bf16* B = KVb + (size_t)nh * DHEAD * DHEAD;

  const v8f vz = {0.f, 0.f, 0.f, 0.f, 0.f, 0.f, 0.f, 0.f};
  v8f acc[4][4];
#pragma unroll
  for (int i = 0; i < 4; i++)
#pragma unroll
    for (int j = 0; j < 4; j++) acc[i][j] = vz;

#pragma unroll
  for (int kb = 0; kb < DHEAD; kb += 32) {
    v16bf a[4], bf[4];
#pragma unroll
    for (int li = 0; li < 4; li++)
      a[li] = load_a16(Q + (size_t)(lt * 64 + li * 16 + l15) * DMODEL + kb + hf * 8);
#pragma unroll
    for (int mi = 0; mi < 4; mi++)
      bf[mi] = load_b16(B + (mi * 16 + l15) * DHEAD + kb + hf * 8);
#pragma unroll
    for (int li = 0; li < 4; li++)
#pragma unroll
      for (int mi = 0; mi < 4; mi++)
        acc[li][mi] = wmma_bf16(a[li], bf[mi], acc[li][mi]);
  }

#pragma unroll
  for (int li = 0; li < 4; li++)
#pragma unroll
    for (int r = 0; r < 8; r++) {
      const int l = lt * 64 + li * 16 + hf * 8 + r;
      const float z = Z[((size_t)n * SEQ + l) * NHEADS + h];
#pragma unroll
      for (int mi = 0; mi < 4; mi++) {
        const int m = mi * 16 + l15;
        Vh[((size_t)n * SEQ + l) * DMODEL + h * DHEAD + m] =
            (__bf16)(acc[li][mi][r] * z);
      }
    }
}

// ---------------------------------------------------------------------------
// Host-side launch
// ---------------------------------------------------------------------------
extern "C" void kernel_launch(void* const* d_in, const int* in_sizes, int n_in,
                              void* d_out, int out_size, void* d_ws,
                              size_t ws_size, hipStream_t stream) {
  (void)in_sizes; (void)n_in; (void)out_size; (void)ws_size;

  const float* queries = (const float*)d_in[0];
  const float* keys    = (const float*)d_in[1];
  const float* values  = (const float*)d_in[2];
  const float* Wq = (const float*)d_in[3];  const float* bq = (const float*)d_in[4];
  const float* Wk = (const float*)d_in[5];  const float* bk = (const float*)d_in[6];
  const float* Wv = (const float*)d_in[7];  const float* bv = (const float*)d_in[8];
  const float* Wo = (const float*)d_in[9];  const float* bo = (const float*)d_in[10];
  float* out = (float*)d_out;

  const size_t NLD = (size_t)NB * SEQ * DMODEL;
  const size_t WEL = (size_t)DMODEL * DMODEL;
  char* w = (char*)d_ws;
  auto alloc = [&](size_t bytes) -> void* {
    void* p = (void*)w;
    w += (bytes + 255) & ~(size_t)255;
    return p;
  };
  __bf16* qf   = (__bf16*)alloc(NLD * 2);
  __bf16* kf   = (__bf16*)alloc(NLD * 2);
  __bf16* vb   = (__bf16*)alloc(NLD * 2);
  __bf16* Vh   = (__bf16*)alloc(NLD * 2);
  __bf16* KVb  = (__bf16*)alloc((size_t)NB * NHEADS * DHEAD * DHEAD * 2);
  float*  ksum = (float*)alloc((size_t)NB * DMODEL * 4);
  float*  Z    = (float*)alloc((size_t)NB * SEQ * NHEADS * 4);
  __bf16* wqb  = (__bf16*)alloc(WEL * 2);
  __bf16* wkb  = (__bf16*)alloc(WEL * 2);
  __bf16* wvb  = (__bf16*)alloc(WEL * 2);
  __bf16* wob  = (__bf16*)alloc(WEL * 2);

  const dim3 blk(256);

  // Weights f32 -> bf16 (one pass each)
  const int cvtBlocks = (int)(WEL / (256 * 4));
  cvt_f32_bf16<<<cvtBlocks, blk, 0, stream>>>(Wq, wqb, (int)WEL);
  cvt_f32_bf16<<<cvtBlocks, blk, 0, stream>>>(Wk, wkb, (int)WEL);
  cvt_f32_bf16<<<cvtBlocks, blk, 0, stream>>>(Wv, wvb, (int)WEL);
  cvt_f32_bf16<<<cvtBlocks, blk, 0, stream>>>(Wo, wob, (int)WEL);

  const int M = NB * SEQ;                                  // 16384
  const int gemmBlocks = (M / 64) * (DMODEL / 512);        // 512

  // Projections (+ elu+1 fused for q,k), bf16 outputs
  gemm_xwT<float, true,  __bf16><<<gemmBlocks, blk, 0, stream>>>(queries, wqb, bq, qf, M, DMODEL, DMODEL);
  gemm_xwT<float, true,  __bf16><<<gemmBlocks, blk, 0, stream>>>(keys,    wkb, bk, kf, M, DMODEL, DMODEL);
  gemm_xwT<float, false, __bf16><<<gemmBlocks, blk, 0, stream>>>(values,  wvb, bv, vb, M, DMODEL, DMODEL);

  // K feature-map column sums
  zero_f32<<<(NB * DMODEL + 255) / 256, blk, 0, stream>>>(ksum, NB * DMODEL);
  ksum_kernel<<<(NB * 16 * DMODEL) / 256, blk, 0, stream>>>(kf, ksum);

  // KV summaries (one block per (n,h))
  kv_kernel<<<NB * NHEADS, dim3(128), 0, stream>>>(kf, vb, KVb);

  // Normalizers
  z_kernel<<<(NB * SEQ * NHEADS) / 256, blk, 0, stream>>>(qf, ksum, Z);

  // Per-head output V = Z * (Q @ KV^T)
  attn_kernel<<<(NB * NHEADS * (SEQ / 64)) / 8, blk, 0, stream>>>(qf, KVb, Z, Vh);

  // Final projection, f32 output
  gemm_xwT<__bf16, false, float><<<gemmBlocks, blk, 0, stream>>>(Vh, wob, bo, out, M, DMODEL, DMODEL);
}
